// UniPoolLayer_32899449487927
// MI455X (gfx1250) — matrix operand order repaired and checked by reading, not compiled, not run on hardware
//
#include <hip/hip_runtime.h>
#include <hip/hip_bf16.h>
#include <math.h>

// Problem constants (match reference)
#define T_TOK 4096
#define DMODEL 2048
#define FFF 8192
#define NEXP 8

// GEMM tile config: 256 threads = 8 waves; block tile 64(M) x 64(N) x 32(K).
// 16 wmma tiles per block; each wave owns 1 m-tile and 2 adjacent n-tiles,
// reusing its A fragment across both wmmas.
#define BM 64
#define BN 64
#define BK 32
#define LSTR 40   // LDS row stride in bf16 elems (80B = 5 x 16B, b128-aligned)

typedef __attribute__((ext_vector_type(16))) __bf16 v16bf;
typedef __attribute__((ext_vector_type(8)))  float  v8f;

// ---- bf16 pack helpers --------------------------------------------------
// Prefer the hardware packed converter; otherwise cheap round-half-up.
#if __has_builtin(__builtin_amdgcn_cvt_pk_bf16_f32)
typedef __attribute__((ext_vector_type(2))) __bf16 v2bf;
__device__ __forceinline__ unsigned pack2bf(float a, float b) {
    v2bf r = __builtin_amdgcn_cvt_pk_bf16_f32(a, b);
    return __builtin_bit_cast(unsigned, r);
}
#else
__device__ __forceinline__ unsigned pack2bf(float a, float b) {
    unsigned ua = __builtin_bit_cast(unsigned, a);
    unsigned ub = __builtin_bit_cast(unsigned, b);
    return ((ua + 0x8000u) >> 16) | ((ub + 0x8000u) & 0xFFFF0000u);
}
#endif
__device__ __forceinline__ unsigned short f2bf_bits(float f) {
    unsigned u = __builtin_bit_cast(unsigned, f);
    return (unsigned short)((u + 0x8000u) >> 16);
}

// ---- routing kernels ----------------------------------------------------
__global__ void init_counts_kernel(int* counts) {
    if (threadIdx.x < NEXP) counts[threadIdx.x] = 0;
}

// one wave (32 lanes) per token: 8 dot products of length D via float4 loads
__global__ void router_kernel(const float* __restrict__ x,
                              const float* __restrict__ Wr,
                              int* __restrict__ idx,
                              float* __restrict__ wgt,
                              int* __restrict__ counts) {
    int wid  = threadIdx.x >> 5;
    int lane = threadIdx.x & 31;
    int t = blockIdx.x * (blockDim.x >> 5) + wid;
    if (t >= T_TOK) return;

    float acc[NEXP];
#pragma unroll
    for (int e = 0; e < NEXP; ++e) acc[e] = 0.f;

    const float4* xr4 = (const float4*)(x + (size_t)t * DMODEL);
    for (int q = lane; q < DMODEL / 4; q += 32) {
        float4 xv = xr4[q];
#pragma unroll
        for (int e = 0; e < NEXP; ++e) {
            float4 wv = ((const float4*)(Wr + (size_t)e * DMODEL))[q];
            acc[e] += xv.x * wv.x + xv.y * wv.y + xv.z * wv.z + xv.w * wv.w;
        }
    }
#pragma unroll
    for (int e = 0; e < NEXP; ++e) {
        for (int off = 16; off > 0; off >>= 1)
            acc[e] += __shfl_xor(acc[e], off, 32);
    }
    if (lane == 0) {
        float mx = acc[0]; int bi = 0;
#pragma unroll
        for (int e = 1; e < NEXP; ++e)
            if (acc[e] > mx) { mx = acc[e]; bi = e; }
        float den = 0.f;
#pragma unroll
        for (int e = 0; e < NEXP; ++e) den += expf(acc[e] - mx);
        idx[t] = bi;
        wgt[t] = 1.0f / den;   // softmax prob of the argmax expert
        atomicAdd(&counts[bi], 1);
    }
}

__global__ void scan_kernel(const int* __restrict__ counts,
                            int* __restrict__ offsets,
                            int* __restrict__ cursors) {
    if (threadIdx.x == 0) {
        int s = 0;
        for (int e = 0; e < NEXP; ++e) {
            offsets[e] = s; cursors[e] = s; s += counts[e];
        }
        offsets[NEXP] = s;
    }
}

__global__ void scatter_kernel(const int* __restrict__ idx,
                               int* __restrict__ cursors,
                               int* __restrict__ perm) {
    int t = blockIdx.x * blockDim.x + threadIdx.x;
    if (t >= T_TOK) return;
    int e = idx[t];
    int slot = atomicAdd(&cursors[e], 1);
    perm[slot] = t;
}

// ---- WMMA fragment loaders (contiguous k order, same for A and B) ---
// A (16x32 bf16): lane row m = lane&15; lanes 0-15 hold K=0..15, lanes 16-31
// hold K=16..31 (contiguous). sA row-major [m][k].
__device__ __forceinline__ v16bf load_a_frag(const __bf16* sA, int mt, int lane) {
    v16bf a;
    int m    = mt * 16 + (lane & 15);
    int koff = (lane >> 4) * 16;
    const __bf16* p0 = &sA[m * LSTR + koff];
#pragma unroll
    for (int i = 0; i < 16; ++i) { a[i] = p0[i]; }
    return a;
}
// B (32x16 bf16): lane col n = lane&15; lanes 0-15 hold K=0..15, lanes 16-31
// hold K=16..31 (contiguous). sBt holds B transposed: sBt[n][k].
__device__ __forceinline__ v16bf load_b_frag(const __bf16* sBt, int nt, int lane) {
    v16bf b;
    int n  = nt * 16 + (lane & 15);
    int kb = (lane >> 4) * 16;
    const __bf16* p = &sBt[n * LSTR + kb];
#pragma unroll
    for (int i = 0; i < 16; ++i) b[i] = p[i];
    return b;
}

// ---- GEMM 1: H[slot, :] = gelu( x[perm[slot], :] @ W1[e] ), bf16 out ----
__global__ __launch_bounds__(256) void moe_ffn1_kernel(
        const float* __restrict__ x, const float* __restrict__ W1,
        const int* __restrict__ perm, const int* __restrict__ offsets,
        unsigned short* __restrict__ H) {
    const int e = blockIdx.z;
    const int seg_lo = offsets[e];
    const int nTok   = offsets[e + 1] - seg_lo;
    const int m0 = blockIdx.x * BM;
    if (m0 >= nTok) return;                       // early-exit overshoot tiles
    const int n0 = blockIdx.y * BN;
    const float* W1e = W1 + (size_t)e * DMODEL * FFF;

    __shared__ __bf16 sA[BM * LSTR];
    __shared__ __bf16 sB[BN * LSTR];
    __shared__ int    sTok[BM];

    const int tid = threadIdx.x;
    const int lane = tid & 31, wid = tid >> 5;
    const int mt = wid & 3;                // m-tile 0..3
    const int nt0 = (wid >> 2) * 2;        // n-tiles {nt0, nt0+1}

    // block-wide gathered-row table (once)
    if (tid < BM) {
        int srow = m0 + tid;
        sTok[tid] = (srow < nTok) ? perm[seg_lo + srow] : -1;
    }
    __syncthreads();

    // per-thread A-staging geometry: 2 float4 quads; q = tid + j*256
    // r = q>>3 (8 quads per 32-wide row), kq = (q&7)*4
    int  rA[2], kqA[2];   const float* pA[2];   bool vA[2];
#pragma unroll
    for (int j = 0; j < 2; ++j) {
        int q = tid + j * 256;
        rA[j]  = q >> 3;
        kqA[j] = (q & 7) * 4;
        int tok = sTok[rA[j]];
        vA[j] = (tok >= 0);
        pA[j] = x + (size_t)(vA[j] ? tok : 0) * DMODEL + kqA[j];
    }
    // per-thread B-staging geometry: 2 column k-quads; q = tid + j*256
    // kg = q>>6 (0..7), n = q&63; loads column n0+n rows k0+kg*4..+3
    int nB[2], kgB[2];   const float* pB[2];
#pragma unroll
    for (int j = 0; j < 2; ++j) {
        int q = tid + j * 256;
        kgB[j] = q >> 6;
        nB[j]  = q & 63;
        pB[j]  = W1e + (size_t)(kgB[j] * 4) * FFF + n0 + nB[j];
    }

    v8f c0 = {}, c1 = {};
    for (int k0 = 0; k0 < DMODEL; k0 += BK) {
        // stage A: b128 gathered loads, packed b64 LDS stores
#pragma unroll
        for (int j = 0; j < 2; ++j) {
            float4 v = make_float4(0.f, 0.f, 0.f, 0.f);
            if (vA[j]) v = *(const float4*)(pA[j] + k0);
            uint2 pkt; pkt.x = pack2bf(v.x, v.y); pkt.y = pack2bf(v.z, v.w);
            *(uint2*)&sA[rA[j] * LSTR + kqA[j]] = pkt;
        }
        // stage B transposed: 4 coalesced b32 column loads, one b64 LDS store
#pragma unroll
        for (int j = 0; j < 2; ++j) {
            const float* p = pB[j] + (size_t)k0 * FFF;
            float f0 = p[0];
            float f1 = p[FFF];
            float f2 = p[2 * FFF];
            float f3 = p[3 * FFF];
            uint2 pkt; pkt.x = pack2bf(f0, f1); pkt.y = pack2bf(f2, f3);
            *(uint2*)&sB[nB[j] * LSTR + kgB[j] * 4] = pkt;
        }
        if (k0 + BK < DMODEL)
            __builtin_prefetch(&W1e[(size_t)(k0 + BK) * FFF + n0], 0, 0);
        __syncthreads();

        v16bf a  = load_a_frag(sA, mt, lane);
        v16bf b0 = load_b_frag(sB, nt0, lane);
        v16bf b1 = load_b_frag(sB, nt0 + 1, lane);
        c0 = __builtin_amdgcn_wmma_f32_16x16x32_bf16(
                 false, a, false, b0, (short)0, c0, false, false);
        c1 = __builtin_amdgcn_wmma_f32_16x16x32_bf16(
                 false, a, false, b1, (short)0, c1, false, false);
        __syncthreads();
    }
    // epilogue: exact gelu, store bf16 H
#pragma unroll
    for (int r = 0; r < 8; ++r) {
        int m = mt * 16 + r + ((lane >> 4) << 3);
        int srow = m0 + m;
        if (srow < nTok) {
            size_t rowb = (size_t)(seg_lo + srow) * FFF;
            float v0 = c0[r];
            float v1 = c1[r];
            float g0 = 0.5f * v0 * (1.0f + erff(v0 * 0.70710678118654752f));
            float g1 = 0.5f * v1 * (1.0f + erff(v1 * 0.70710678118654752f));
            H[rowb + n0 + nt0 * 16 + (lane & 15)]       = f2bf_bits(g0);
            H[rowb + n0 + (nt0 + 1) * 16 + (lane & 15)] = f2bf_bits(g1);
        }
    }
}

// ---- GEMM 2: out[tok, :] = w[tok] * ( H[slot, :] @ W2[e] ) --------------
__global__ __launch_bounds__(256) void moe_ffn2_kernel(
        const unsigned short* __restrict__ H, const float* __restrict__ W2,
        const int* __restrict__ perm, const int* __restrict__ offsets,
        const float* __restrict__ wgt, float* __restrict__ out) {
    const int e = blockIdx.z;
    const int seg_lo = offsets[e];
    const int nTok   = offsets[e + 1] - seg_lo;
    const int m0 = blockIdx.x * BM;
    if (m0 >= nTok) return;
    const int n0 = blockIdx.y * BN;
    const float* W2e = W2 + (size_t)e * FFF * DMODEL;

    __shared__ __bf16 sA[BM * LSTR];
    __shared__ __bf16 sB[BN * LSTR];

    const int tid = threadIdx.x;
    const int lane = tid & 31, wid = tid >> 5;
    const int mt = wid & 3;
    const int nt0 = (wid >> 2) * 2;

    // A-staging geometry: H rows are contiguous bf16 slots; one b128 load
    // (8 bf16) per thread: r = tid>>2, k8 = (tid&3)*8
    const int rA  = tid >> 2;
    const int k8A = (tid & 3) * 8;
    const bool vAr = (m0 + rA) < nTok;
    const unsigned short* pA =
        H + (size_t)(seg_lo + (vAr ? (m0 + rA) : 0)) * FFF + k8A;

    int nB[2], kgB[2];   const float* pB[2];
#pragma unroll
    for (int j = 0; j < 2; ++j) {
        int q = tid + j * 256;
        kgB[j] = q >> 6;
        nB[j]  = q & 63;
        pB[j]  = W2e + (size_t)(kgB[j] * 4) * DMODEL + n0 + nB[j];
    }

    v8f c0 = {}, c1 = {};
    for (int k0 = 0; k0 < FFF; k0 += BK) {
        // stage A (already bf16): one b128 load + b128 LDS store
        {
            uint4 v = make_uint4(0u, 0u, 0u, 0u);
            if (vAr) v = *(const uint4*)(pA + k0);
            *(uint4*)&sA[rA * LSTR + k8A] = v;
        }
        // stage B transposed
#pragma unroll
        for (int j = 0; j < 2; ++j) {
            const float* p = pB[j] + (size_t)k0 * DMODEL;
            float f0 = p[0];
            float f1 = p[DMODEL];
            float f2 = p[2 * DMODEL];
            float f3 = p[3 * DMODEL];
            uint2 pkt; pkt.x = pack2bf(f0, f1); pkt.y = pack2bf(f2, f3);
            *(uint2*)&sB[nB[j] * LSTR + kgB[j] * 4] = pkt;
        }
        if (k0 + BK < FFF)
            __builtin_prefetch(&W2e[(size_t)(k0 + BK) * DMODEL + n0], 0, 0);
        __syncthreads();

        v16bf a  = load_a_frag(sA, mt, lane);
        v16bf b0 = load_b_frag(sB, nt0, lane);
        v16bf b1 = load_b_frag(sB, nt0 + 1, lane);
        c0 = __builtin_amdgcn_wmma_f32_16x16x32_bf16(
                 false, a, false, b0, (short)0, c0, false, false);
        c1 = __builtin_amdgcn_wmma_f32_16x16x32_bf16(
                 false, a, false, b1, (short)0, c1, false, false);
        __syncthreads();
    }
    // epilogue: scale by router prob, scatter to original token row
#pragma unroll
    for (int r = 0; r < 8; ++r) {
        int m = mt * 16 + r + ((lane >> 4) << 3);
        int srow = m0 + m;
        if (srow < nTok) {
            int tok = perm[seg_lo + srow];
            float wv = wgt[tok];
            size_t rowb = (size_t)tok * DMODEL;
            out[rowb + n0 + nt0 * 16 + (lane & 15)]       = c0[r] * wv;
            out[rowb + n0 + (nt0 + 1) * 16 + (lane & 15)] = c1[r] * wv;
        }
    }
}

// ---- host launcher ------------------------------------------------------
extern "C" void kernel_launch(void* const* d_in, const int* in_sizes, int n_in,
                              void* d_out, int out_size, void* d_ws, size_t ws_size,
                              hipStream_t stream) {
    (void)in_sizes; (void)n_in; (void)out_size; (void)ws_size;
    const float* x  = (const float*)d_in[0];   // [T, D]
    const float* Wr = (const float*)d_in[1];   // [E, D]
    const float* W1 = (const float*)d_in[2];   // [E, D, F]
    const float* W2 = (const float*)d_in[3];   // [E, F, D]
    float* out = (float*)d_out;                // [T, D]

    char* ws = (char*)d_ws;
    int*   counts  = (int*)(ws + 0);
    int*   offsets = (int*)(ws + 64);
    int*   cursors = (int*)(ws + 128);
    int*   idx     = (int*)(ws + 256);
    float* wgt     = (float*)(ws + 256 + 4 * T_TOK);
    int*   perm    = (int*)(ws + 256 + 8 * T_TOK);
    unsigned short* H = (unsigned short*)(ws + 65536);   // T*F bf16 = 64 MB

    init_counts_kernel<<<1, 32, 0, stream>>>(counts);
    router_kernel<<<T_TOK / 8, 256, 0, stream>>>(x, Wr, idx, wgt, counts);
    scan_kernel<<<1, 32, 0, stream>>>(counts, offsets, cursors);
    scatter_kernel<<<T_TOK / 256, 256, 0, stream>>>(idx, cursors, perm);
    moe_ffn1_kernel<<<dim3(T_TOK / BM, FFF / BN, NEXP), 256, 0, stream>>>(
        x, W1, perm, offsets, H);
    moe_ffn2_kernel<<<dim3(T_TOK / BM, DMODEL / BN, NEXP), 256, 0, stream>>>(
        H, W2, perm, offsets, wgt, out);
}
